// AverageMeshNetworkPEARUR_86303072845951
// MI455X (gfx1250) — matrix operand order repaired and checked by reading, not compiled, not run on hardware
//
#include <hip/hip_runtime.h>

// ---------------- problem constants (from reference setup) ----------------
#define NP   16384   // spider patches
#define S    25      // nodes per patch
#define EP   72      // edges per patch (shared topology)
#define F    64      // in_dim
#define H    128     // 2*in_dim
#define HQ   32      // H/4
#define RD   128     // readout dim
#define RCAT 224     // F + H + HQ
#define BB   4       // mesh graphs
#define PM   4096    // mesh nodes per graph
#define EM   65536   // mesh edges
#define MH   128
#define URH  256
#define URO  32
#define OUTN 50
#define NEGS 0.01f
#define EPSV 1e-5f

typedef __attribute__((ext_vector_type(16))) _Float16 v16h;
typedef __attribute__((ext_vector_type(8)))  float    v8f;

union F16x16 { v16h v; _Float16 h[16]; };
union F32x8  { v8f  v; float    f[8];  };

__device__ __forceinline__ float lrelu(float x) { return x >= 0.f ? x : NEGS * x; }

// A fragment (16x32 f16, row-major, leading dim ldk) per CDNA5 ISA layout:
// lanes 0-15: elems 0-7 = K kb..kb+7, elems 8-15 = K kb+16..kb+23
// lanes 16-31: elems 0-7 = K kb+8..kb+15, elems 8-15 = K kb+24..kb+31
__device__ __forceinline__ v16h ldA(const _Float16* a, int ldk, int mBase, int kBase, int lane) {
  int half = lane >> 4, r = lane & 15;
  const _Float16* p = a + (size_t)(mBase + r) * ldk + kBase + half * 8;
  F16x16 t;
#pragma unroll
  for (int i = 0; i < 8; ++i) { t.h[i] = p[i]; t.h[8 + i] = p[16 + i]; }
  return t.v;
}

// B fragment (32x16 f16) from weights stored TRANSPOSED [N][K]:
// lane holds column N=lane%16, elems 0-7 = K kBase+8*(lane/16)+j, elems 8-15 = K kBase+16+8*(lane/16)+(j-8)
__device__ __forceinline__ v16h ldB(const _Float16* wt, int ldk, int nBase, int kBase, int lane) {
  int half = lane >> 4, r = lane & 15;
  const _Float16* p = wt + (size_t)(nBase + r) * ldk + kBase + half * 8;
  F16x16 t;
#pragma unroll
  for (int i = 0; i < 8; ++i) { t.h[i] = p[i]; t.h[8 + i] = p[16 + i]; }
  return t.v;
}

#define WMMA_F16(a, b, c) __builtin_amdgcn_wmma_f32_16x16x32_f16(false, (a), false, (b), (short)0, (c), false, false)

// ---------------- tiny prep kernels ----------------

// W [K][N] f32 -> out [N][K] f16 (transposed, for B-fragment loads)
__global__ void prep_wt(const float* __restrict__ W, _Float16* __restrict__ out, int K, int N) {
  int i = blockIdx.x * blockDim.x + threadIdx.x;
  if (i >= K * N) return;
  int n = i / K, k = i % K;
  out[n * K + k] = (_Float16)W[(size_t)k * N + n];
}

__global__ void patch_deg(const int* __restrict__ esrc, const int* __restrict__ edst,
                          float* __restrict__ nsp, float* __restrict__ ndp) {
  int i = threadIdx.x;
  if (i >= 32) return;
  int co = 0, ci = 0;
  for (int e = 0; e < EP; ++e) { co += (esrc[e] == i); ci += (edst[e] == i); }
  nsp[i] = (i < S) ? rsqrtf(fmaxf((float)co, 1.f)) : 0.f;
  ndp[i] = (i < S) ? rsqrtf(fmaxf((float)ci, 1.f)) : 0.f;
}

// ---------------- patch stage ----------------

// per-patch: graph aggregation (25x64) + WMMA matmul with Wp1 -> h1_pre [NP,25,128]; also r0
__global__ void patch_conv1(const float* __restrict__ x, const float* __restrict__ ew,
                            const int* __restrict__ esrc, const int* __restrict__ edst,
                            const float* __restrict__ nsp, const float* __restrict__ ndp,
                            const _Float16* __restrict__ wt1,   // [H][F] transposed f16
                            float* __restrict__ h1_pre, float* __restrict__ rcat) {
  __shared__ float xs[S * F];
  __shared__ float aggf[S * F];
  __shared__ alignas(16) _Float16 aggH[32 * F];
  int p = blockIdx.x, tid = threadIdx.x;
  for (int i = tid; i < S * F; i += 128) { xs[i] = x[(size_t)p * S * F + i]; aggf[i] = 0.f; }
  __syncthreads();
  if (tid < F) {                       // r0 = mean over nodes
    float s = 0.f;
    for (int r = 0; r < S; ++r) s += xs[r * F + tid];
    rcat[(size_t)p * RCAT + tid] = s * (1.f / S);
  }
  if (tid < F) {                       // aggregation: thread owns feature tid, serial edges
    for (int e = 0; e < EP; ++e) {
      int sn = esrc[e], dn = edst[e];
      float c = nsp[sn] * ew[(size_t)p * EP + e];
      aggf[dn * F + tid] += c * xs[sn * F + tid];
    }
  }
  __syncthreads();
  for (int i = tid; i < 32 * F; i += 128) {
    int r = i / F;
    aggH[i] = (_Float16)((r < S) ? aggf[i] * ndp[r] : 0.f);
  }
  __syncthreads();
  int wave = tid >> 5, lane = tid & 31;
  for (int nt = wave; nt < 8; nt += 4) {
    for (int mt = 0; mt < 2; ++mt) {
      v8f acc = {0.f, 0.f, 0.f, 0.f, 0.f, 0.f, 0.f, 0.f};
#pragma unroll
      for (int kk = 0; kk < F; kk += 32)
        acc = WMMA_F16(ldA(aggH, F, mt * 16, kk, lane), ldB(wt1, F, nt * 16, kk, lane), acc);
      F32x8 d; d.v = acc;
      int col = nt * 16 + (lane & 15);
      int mb = mt * 16 + (lane >> 4) * 8;
#pragma unroll
      for (int r = 0; r < 8; ++r) {
        int m = mb + r;
        if (m < S) h1_pre[((size_t)p * S + m) * H + col] = d.f[r];
      }
    }
  }
}

// per-channel sum/sumsq over rows, grouped (G groups of rowsPerGroup rows); blockDim == C
__global__ void chan_stats(const float* __restrict__ x, float* __restrict__ sums,
                           int C, int rowsPerBlock, int rowsPerGroup) {
  int c = threadIdx.x;
  long long start = (long long)blockIdx.x * rowsPerBlock;
  int g = (int)(start / rowsPerGroup);
  float s = 0.f, s2 = 0.f;
  for (int r = 0; r < rowsPerBlock; ++r) {
    float v = x[(start + r) * (long long)C + c];
    s += v; s2 += v * v;
  }
  atomicAdd(&sums[(size_t)g * 2 * C + c], s);
  atomicAdd(&sums[(size_t)g * 2 * C + C + c], s2);
}

// ab[g][0][c] = gamma*rstd ; ab[g][1][c] = beta - mean*gamma*rstd
__global__ void stats_fin(const float* __restrict__ sums, const float* __restrict__ gg,
                          const float* __restrict__ bb, float* __restrict__ ab,
                          int C, int G, float invN) {
  int i = blockIdx.x * blockDim.x + threadIdx.x;
  if (i >= G * C) return;
  int g = i / C, c = i % C;
  float m = sums[g * 2 * C + c] * invN;
  float v = sums[g * 2 * C + C + c] * invN - m * m;
  float a = gg[c] * rsqrtf(v + EPSV);
  ab[g * 2 * C + c] = a;
  ab[g * 2 * C + C + c] = bb[c] - m * a;
}

// per-patch: normalize+lrelu h1, r1, aggregate, WMMA with Wp2 -> h2_pre [NP,25,32]
__global__ void patch_conv2(const float* __restrict__ h1_pre, const float* __restrict__ ew,
                            const int* __restrict__ esrc, const int* __restrict__ edst,
                            const float* __restrict__ nsp, const float* __restrict__ ndp,
                            const float* __restrict__ ab1,
                            const _Float16* __restrict__ wt2,   // [HQ][H] transposed f16
                            float* __restrict__ h2_pre, float* __restrict__ rcat) {
  __shared__ float h1n[S * H];
  __shared__ float aggf[S * H];
  __shared__ alignas(16) _Float16 aggH[32 * H];
  int p = blockIdx.x, tid = threadIdx.x;          // 128 threads == H channels
  float a = ab1[tid], b = ab1[H + tid];
  for (int r = 0; r < S; ++r) {
    float v = h1_pre[((size_t)p * S + r) * H + tid];
    h1n[r * H + tid] = lrelu(a * v + b);
    aggf[r * H + tid] = 0.f;
  }
  { float s = 0.f;
    for (int r = 0; r < S; ++r) s += h1n[r * H + tid];
    rcat[(size_t)p * RCAT + F + tid] = s * (1.f / S); }
  for (int e = 0; e < EP; ++e) {                  // thread owns channel tid
    int sn = esrc[e], dn = edst[e];
    float c = nsp[sn] * ew[(size_t)p * EP + e];
    aggf[dn * H + tid] += c * h1n[sn * H + tid];
  }
  __syncthreads();
  for (int i = tid; i < 32 * H; i += 128) {
    int r = i / H;
    aggH[i] = (_Float16)((r < S) ? aggf[i] * ndp[r] : 0.f);
  }
  __syncthreads();
  int wave = tid >> 5, lane = tid & 31;
  int mt = wave >> 1, nt = wave & 1;
  v8f acc = {0.f, 0.f, 0.f, 0.f, 0.f, 0.f, 0.f, 0.f};
#pragma unroll
  for (int kk = 0; kk < H; kk += 32)
    acc = WMMA_F16(ldA(aggH, H, mt * 16, kk, lane), ldB(wt2, H, nt * 16, kk, lane), acc);
  F32x8 d; d.v = acc;
  int col = nt * 16 + (lane & 15);
  int mb = mt * 16 + (lane >> 4) * 8;
#pragma unroll
  for (int r = 0; r < 8; ++r) {
    int m = mb + r;
    if (m < S) h2_pre[((size_t)p * S + m) * HQ + col] = d.f[r];
  }
}

__global__ void r2_kernel(const float* __restrict__ h2_pre, const float* __restrict__ ab2,
                          float* __restrict__ rcat) {
  int i = blockIdx.x * blockDim.x + threadIdx.x;
  if (i >= NP * HQ) return;
  int p = i / HQ, c = i % HQ;
  float a = ab2[c], b = ab2[HQ + c], s = 0.f;
  for (int r = 0; r < S; ++r) s += lrelu(a * h2_pre[((size_t)p * S + r) * HQ + c] + b);
  rcat[(size_t)p * RCAT + F + H + c] = s * (1.f / S);
}

// 16 patches/block: [16,224]@[224,128] WMMA + per-row instance norm + lrelu -> readouts
__global__ void embed_kernel(const float* __restrict__ rcat, const _Float16* __restrict__ wtE, // [RD][RCAT]
                             float* __restrict__ readouts) {
  __shared__ alignas(16) _Float16 aT[16 * RCAT];
  __shared__ float embS[16 * RD];
  __shared__ float mS[16], rS[16];
  int p0 = blockIdx.x * 16, tid = threadIdx.x;
  for (int i = tid; i < 16 * RCAT; i += 128) aT[i] = (_Float16)rcat[(size_t)p0 * RCAT + i];
  __syncthreads();
  int wave = tid >> 5, lane = tid & 31;
  for (int nt = wave; nt < 8; nt += 4) {
    v8f acc = {0.f, 0.f, 0.f, 0.f, 0.f, 0.f, 0.f, 0.f};
#pragma unroll
    for (int kk = 0; kk < RCAT; kk += 32)
      acc = WMMA_F16(ldA(aT, RCAT, 0, kk, lane), ldB(wtE, RCAT, nt * 16, kk, lane), acc);
    F32x8 d; d.v = acc;
    int col = nt * 16 + (lane & 15), mb = (lane >> 4) * 8;
#pragma unroll
    for (int r = 0; r < 8; ++r) embS[(mb + r) * RD + col] = d.f[r];
  }
  __syncthreads();
  if (tid < 16) {
    float s = 0.f, s2 = 0.f;
    for (int c = 0; c < RD; ++c) { float v = embS[tid * RD + c]; s += v; s2 += v * v; }
    float m = s * (1.f / RD);
    mS[tid] = m;
    rS[tid] = rsqrtf(s2 * (1.f / RD) - m * m + EPSV);
  }
  __syncthreads();
  for (int i = tid; i < 16 * RD; i += 128) {
    int r = i / RD;
    readouts[(size_t)p0 * RD + i] = lrelu((embS[i] - mS[r]) * rS[r]);
  }
}

// ---------------- mesh stage ----------------

__global__ void mesh_deg(const int* __restrict__ msrc, const int* __restrict__ mdst,
                         float* __restrict__ degO, float* __restrict__ degI) {
  int e = blockIdx.x * blockDim.x + threadIdx.x;
  if (e >= EM) return;
  atomicAdd(&degO[msrc[e]], 1.f);
  atomicAdd(&degI[mdst[e]], 1.f);
}
__global__ void mesh_deg_fin(float* __restrict__ degO, float* __restrict__ degI) {
  int i = blockIdx.x * blockDim.x + threadIdx.x;
  if (i >= PM) return;
  degO[i] = rsqrtf(fmaxf(degO[i], 1.f));
  degI[i] = rsqrtf(fmaxf(degI[i], 1.f));
}

// scatter aggregation: agg[b,dst,:] += ns[src]*ew[b,e]*feat(b,src,:)
// mode 0: feat = readouts; mode 1: feat = lrelu(a*m_pre+b)
__global__ void mesh_agg(const float* __restrict__ feats, const float* __restrict__ ab,
                         const float* __restrict__ mew,
                         const int* __restrict__ msrc, const int* __restrict__ mdst,
                         const float* __restrict__ nsm, float* __restrict__ agg, int mode) {
  int b = blockIdx.y, tid = threadIdx.x;          // 128 threads == MH channels
  int e0 = blockIdx.x * 32;
  float a = 1.f, sh = 0.f;
  if (mode) { a = ab[b * 2 * MH + tid]; sh = ab[b * 2 * MH + MH + tid]; }
  for (int k = 0; k < 32; ++k) {
    int e = e0 + k;
    int sn = msrc[e], dn = mdst[e];
    float c = nsm[sn] * mew[(size_t)b * EM + e];
    float v = feats[((size_t)b * PM + sn) * MH + tid];
    if (mode) v = lrelu(a * v + sh);
    atomicAdd(&agg[((size_t)b * PM + dn) * MH + tid], c * v);
  }
}

// (nd[i]*agg[row]) @ W[128x128] -> out ; 32 rows/block
__global__ void mesh_mm(const float* __restrict__ agg, const float* __restrict__ ndm,
                        const _Float16* __restrict__ wt,   // [MH][MH] transposed f16
                        float* __restrict__ out) {
  __shared__ alignas(16) _Float16 aT[32 * MH];
  int r0 = blockIdx.x * 32, tid = threadIdx.x;
  for (int i = tid; i < 32 * MH; i += 128) {
    int row = r0 + i / MH;
    aT[i] = (_Float16)(agg[(size_t)row * MH + (i % MH)] * ndm[row & (PM - 1)]);
  }
  __syncthreads();
  int wave = tid >> 5, lane = tid & 31;
  for (int nt = wave; nt < 8; nt += 4)
    for (int mt = 0; mt < 2; ++mt) {
      v8f acc = {0.f, 0.f, 0.f, 0.f, 0.f, 0.f, 0.f, 0.f};
#pragma unroll
      for (int kk = 0; kk < MH; kk += 32)
        acc = WMMA_F16(ldA(aT, MH, mt * 16, kk, lane), ldB(wt, MH, nt * 16, kk, lane), acc);
      F32x8 d; d.v = acc;
      int col = nt * 16 + (lane & 15), m = r0 + mt * 16 + (lane >> 4) * 8;
#pragma unroll
      for (int r = 0; r < 8; ++r) out[(size_t)(m + r) * MH + col] = d.f[r];
    }
}

// y = lrelu( lrelu(a*m_pre+b) @ phi[128x256] ); accumulate column sums per batch
__global__ void phi_kernel(const float* __restrict__ m_pre, const float* __restrict__ ab,
                           const _Float16* __restrict__ wtphi,   // [URH][MH] transposed f16
                           float* __restrict__ accum) {          // [BB][URH]
  __shared__ alignas(16) _Float16 aT[32 * MH];
  int r0 = blockIdx.x * 32, tid = threadIdx.x;
  int b = r0 / PM;
  for (int i = tid; i < 32 * MH; i += 128) {
    int c = i % MH;
    aT[i] = (_Float16)lrelu(ab[b * 2 * MH + c] * m_pre[(size_t)r0 * MH + i] + ab[b * 2 * MH + MH + c]);
  }
  __syncthreads();
  int wave = tid >> 5, lane = tid & 31;
  for (int nt = wave; nt < 16; nt += 4)
    for (int mt = 0; mt < 2; ++mt) {
      v8f acc = {0.f, 0.f, 0.f, 0.f, 0.f, 0.f, 0.f, 0.f};
#pragma unroll
      for (int kk = 0; kk < MH; kk += 32)
        acc = WMMA_F16(ldA(aT, MH, mt * 16, kk, lane), ldB(wtphi, MH, nt * 16, kk, lane), acc);
      F32x8 d; d.v = acc;
      float s = 0.f;
#pragma unroll
      for (int r = 0; r < 8; ++r) s += lrelu(d.f[r]);
      atomicAdd(&accum[b * URH + nt * 16 + (lane & 15)], s);
    }
}

// u[b,j] = lrelu( (accum[b,:]/PM) @ rho ) ; 128 threads = BB*URO
__global__ void readout_fin(const float* __restrict__ accum, const float* __restrict__ rho, // [URH][URO]
                            float* __restrict__ ubuf, int col0) {
  int i = threadIdx.x;
  int b = i / URO, j = i % URO;
  float s = 0.f;
  for (int k = 0; k < URH; ++k) s += (accum[b * URH + k] * (1.f / PM)) * rho[k * URO + j];
  ubuf[b * (2 * URO) + col0 + j] = lrelu(s);
}

// out = lrelu(ubuf) @ Wc [256,50]
__global__ void classifier(const float* __restrict__ ubuf, const float* __restrict__ Wc,
                           float* __restrict__ out) {
  int j = threadIdx.x;
  if (j >= OUTN) return;
  float s = 0.f;
  for (int k = 0; k < 2 * URO * BB; ++k) s += lrelu(ubuf[k]) * Wc[k * OUTN + j];
  out[j] = s;
}

// ---------------- launch ----------------
// Workspace requirement: ~310 MB (h1_pre dominates at 210 MB).
extern "C" void kernel_launch(void* const* d_in, const int* in_sizes, int n_in,
                              void* d_out, int out_size, void* d_ws, size_t ws_size,
                              hipStream_t stream) {
  (void)in_sizes; (void)n_in; (void)out_size; (void)ws_size;
  const float* patch_feats = (const float*)d_in[0];
  const float* patch_ew    = (const float*)d_in[1];
  const int*   patch_src   = (const int*)d_in[2];
  const int*   patch_dst   = (const int*)d_in[3];
  const int*   mesh_src    = (const int*)d_in[4];
  const int*   mesh_dst    = (const int*)d_in[5];
  const float* mesh_ew     = (const float*)d_in[6];
  const float* Wp1 = (const float*)d_in[7];
  const float* Wp2 = (const float*)d_in[8];
  const float* gnp1_g = (const float*)d_in[9],  *gnp1_b = (const float*)d_in[10];
  const float* gnp2_g = (const float*)d_in[11], *gnp2_b = (const float*)d_in[12];
  const float* W_embed = (const float*)d_in[13];
  const float* Wm1 = (const float*)d_in[14], *Wm2 = (const float*)d_in[15];
  const float* gnm1_g = (const float*)d_in[16], *gnm1_b = (const float*)d_in[17];
  const float* gnm2_g = (const float*)d_in[18], *gnm2_b = (const float*)d_in[19];
  const float* ur1_phi = (const float*)d_in[20], *ur1_rho = (const float*)d_in[21];
  const float* ur2_phi = (const float*)d_in[22], *ur2_rho = (const float*)d_in[23];
  const float* Wc = (const float*)d_in[24];

  char* wsb = (char*)d_ws;
  size_t cur = 0;
  auto alloc = [&](size_t bytes) -> void* {
    void* p = wsb + cur;
    cur = (cur + bytes + 255) & ~(size_t)255;
    return p;
  };
  float* nsp    = (float*)alloc(32 * 4);
  float* ndp    = (float*)alloc(32 * 4);
  float* stats1 = (float*)alloc(2 * H * 4);
  float* ab1    = (float*)alloc(2 * H * 4);
  float* stats2 = (float*)alloc(2 * HQ * 4);
  float* ab2    = (float*)alloc(2 * HQ * 4);
  float* nsm    = (float*)alloc(PM * 4);
  float* ndm    = (float*)alloc(PM * 4);
  float* statsM = (float*)alloc(BB * 2 * MH * 4);
  float* abM1   = (float*)alloc(BB * 2 * MH * 4);
  float* abM2   = (float*)alloc(BB * 2 * MH * 4);
  float* accum1 = (float*)alloc(BB * URH * 4);
  float* accum2 = (float*)alloc(BB * URH * 4);
  float* ubuf   = (float*)alloc(BB * 2 * URO * 4);
  _Float16* wtP1 = (_Float16*)alloc((size_t)H * F * 2);
  _Float16* wtP2 = (_Float16*)alloc((size_t)HQ * H * 2);
  _Float16* wtE  = (_Float16*)alloc((size_t)RD * RCAT * 2);
  _Float16* wtM1 = (_Float16*)alloc((size_t)MH * MH * 2);
  _Float16* wtM2 = (_Float16*)alloc((size_t)MH * MH * 2);
  _Float16* wtF1 = (_Float16*)alloc((size_t)URH * MH * 2);
  _Float16* wtF2 = (_Float16*)alloc((size_t)URH * MH * 2);
  float* rcat     = (float*)alloc((size_t)NP * RCAT * 4);
  float* readouts = (float*)alloc((size_t)NP * RD * 4);
  float* aggM     = (float*)alloc((size_t)BB * PM * MH * 4);
  float* m1pre    = (float*)alloc((size_t)BB * PM * MH * 4);
  float* m2pre    = (float*)alloc((size_t)BB * PM * MH * 4);
  float* h2pre    = (float*)alloc((size_t)NP * S * HQ * 4);
  float* h1pre    = (float*)alloc((size_t)NP * S * H * 4);

  // weight prep (f32 -> transposed f16)
  patch_deg<<<1, 32, 0, stream>>>(patch_src, patch_dst, nsp, ndp);
  prep_wt<<<(F * H + 255) / 256, 256, 0, stream>>>(Wp1, wtP1, F, H);
  prep_wt<<<(H * HQ + 255) / 256, 256, 0, stream>>>(Wp2, wtP2, H, HQ);
  prep_wt<<<(RCAT * RD + 255) / 256, 256, 0, stream>>>(W_embed, wtE, RCAT, RD);
  prep_wt<<<(RD * MH + 255) / 256, 256, 0, stream>>>(Wm1, wtM1, RD, MH);
  prep_wt<<<(MH * MH + 255) / 256, 256, 0, stream>>>(Wm2, wtM2, MH, MH);
  prep_wt<<<(MH * URH + 255) / 256, 256, 0, stream>>>(ur1_phi, wtF1, MH, URH);
  prep_wt<<<(MH * URH + 255) / 256, 256, 0, stream>>>(ur2_phi, wtF2, MH, URH);

  // patch stage
  patch_conv1<<<NP, 128, 0, stream>>>(patch_feats, patch_ew, patch_src, patch_dst,
                                      nsp, ndp, wtP1, h1pre, rcat);
  hipMemsetAsync(stats1, 0, 2 * H * 4, stream);
  chan_stats<<<800, H, 0, stream>>>(h1pre, stats1, H, 512, NP * S);
  stats_fin<<<1, H, 0, stream>>>(stats1, gnp1_g, gnp1_b, ab1, H, 1, 1.f / (NP * S));
  patch_conv2<<<NP, 128, 0, stream>>>(h1pre, patch_ew, patch_src, patch_dst,
                                      nsp, ndp, ab1, wtP2, h2pre, rcat);
  hipMemsetAsync(stats2, 0, 2 * HQ * 4, stream);
  chan_stats<<<800, HQ, 0, stream>>>(h2pre, stats2, HQ, 512, NP * S);
  stats_fin<<<1, HQ, 0, stream>>>(stats2, gnp2_g, gnp2_b, ab2, HQ, 1, 1.f / (NP * S));
  r2_kernel<<<(NP * HQ) / 128, 128, 0, stream>>>(h2pre, ab2, rcat);
  embed_kernel<<<NP / 16, 128, 0, stream>>>(rcat, wtE, readouts);

  // mesh degrees
  hipMemsetAsync(nsm, 0, PM * 4, stream);
  hipMemsetAsync(ndm, 0, PM * 4, stream);
  mesh_deg<<<EM / 256, 256, 0, stream>>>(mesh_src, mesh_dst, nsm, ndm);
  mesh_deg_fin<<<PM / 128, 128, 0, stream>>>(nsm, ndm);

  // mesh conv1 + readout1
  hipMemsetAsync(aggM, 0, (size_t)BB * PM * MH * 4, stream);
  { dim3 g(EM / 32, BB);
    mesh_agg<<<g, MH, 0, stream>>>(readouts, nullptr, mesh_ew, mesh_src, mesh_dst, nsm, aggM, 0); }
  mesh_mm<<<(BB * PM) / 32, 128, 0, stream>>>(aggM, ndm, wtM1, m1pre);
  hipMemsetAsync(statsM, 0, BB * 2 * MH * 4, stream);
  chan_stats<<<(BB * PM) / 256, MH, 0, stream>>>(m1pre, statsM, MH, 256, PM);
  stats_fin<<<(BB * MH) / 128, 128, 0, stream>>>(statsM, gnm1_g, gnm1_b, abM1, MH, BB, 1.f / PM);
  hipMemsetAsync(accum1, 0, BB * URH * 4, stream);
  phi_kernel<<<(BB * PM) / 32, 128, 0, stream>>>(m1pre, abM1, wtF1, accum1);
  readout_fin<<<1, BB * URO, 0, stream>>>(accum1, ur1_rho, ubuf, 0);

  // mesh conv2 + readout2
  hipMemsetAsync(aggM, 0, (size_t)BB * PM * MH * 4, stream);
  { dim3 g(EM / 32, BB);
    mesh_agg<<<g, MH, 0, stream>>>(m1pre, abM1, mesh_ew, mesh_src, mesh_dst, nsm, aggM, 1); }
  mesh_mm<<<(BB * PM) / 32, 128, 0, stream>>>(aggM, ndm, wtM2, m2pre);
  hipMemsetAsync(statsM, 0, BB * 2 * MH * 4, stream);
  chan_stats<<<(BB * PM) / 256, MH, 0, stream>>>(m2pre, statsM, MH, 256, PM);
  stats_fin<<<(BB * MH) / 128, 128, 0, stream>>>(statsM, gnm2_g, gnm2_b, abM2, MH, BB, 1.f / PM);
  hipMemsetAsync(accum2, 0, BB * URH * 4, stream);
  phi_kernel<<<(BB * PM) / 32, 128, 0, stream>>>(m2pre, abM2, wtF2, accum2);
  readout_fin<<<1, BB * URO, 0, stream>>>(accum2, ur2_rho, ubuf, URO);

  classifier<<<1, 64, 0, stream>>>(ubuf, Wc, (float*)d_out);
}
